// DeepHandwritingSynthesisModel_54855322305219
// MI455X (gfx1250) — matrix operand order repaired and checked by reading, not compiled, not run on hardware
//
#include <hip/hip_runtime.h>
#include <hip/hip_bf16.h>

typedef __attribute__((ext_vector_type(16))) _Float16 v16h;
typedef __attribute__((ext_vector_type(8)))  _Float16 h8;
typedef __attribute__((ext_vector_type(8)))  float    v8f;

// ---------------- model dims ----------------
#define BATCH 32
#define TSTEPS 400
#define UNITS 400
#define GATES 1600          // 4*UNITS
#define KATT 10
#define NCHARS 73
#define UC 50
#define NMIX 20
#define YCOLS 121           // 6*NMIX+1

#define K1PAD 480           // 3 + 73 + 400 -> pad 480  (15 ktiles)
#define K23PAD 896          // 3 + 73 + 400 + 400 -> pad 896 (28 ktiles)
#define KT1 15
#define KT23 28
#define NT_G 100            // 1600/16
#define HROWS (BATCH*TSTEPS) // 12800
#define KMDNPAD 416         // 400 -> pad 416 (13 ktiles)
#define KTM 13
#define NTM 8               // 128 cols pad

#define NWG 13
#define NTHR 256
#define TOTTHR (NWG*NTHR)

// ---------------- workspace layout (bytes) ----------------
static constexpr size_t W1P_OFF = 0;
static constexpr size_t W1P_SZ  = (size_t)KT1 * NT_G * 512 * 2;   // 1,536,000
static constexpr size_t W2P_OFF = W1P_OFF + W1P_SZ;
static constexpr size_t W2P_SZ  = (size_t)KT23 * NT_G * 512 * 2;  // 2,867,200
static constexpr size_t W3P_OFF = W2P_OFF + W2P_SZ;
static constexpr size_t W3P_SZ  = W2P_SZ;
static constexpr size_t WMP_OFF = W3P_OFF + W3P_SZ;
static constexpr size_t WMP_SZ  = (size_t)KTM * NTM * 512 * 2;    // 106,496
static constexpr size_t ZERO_OFF = WMP_OFF + WMP_SZ;
static constexpr size_t A1_OFF = ZERO_OFF;                 static constexpr size_t A1_SZ = (size_t)BATCH * K1PAD * 2;
static constexpr size_t A2_OFF = A1_OFF + A1_SZ;           static constexpr size_t A2_SZ = (size_t)BATCH * K23PAD * 2;
static constexpr size_t A3_OFF = A2_OFF + A2_SZ;           static constexpr size_t A3_SZ = A2_SZ;
static constexpr size_t Z_OFF  = A3_OFF + A3_SZ;           static constexpr size_t Z_SZ  = (size_t)BATCH * GATES * 4;
static constexpr size_t C1_OFF = Z_OFF + Z_SZ;             static constexpr size_t C_SZ  = (size_t)BATCH * UNITS * 4;
static constexpr size_t C2_OFF = C1_OFF + C_SZ;
static constexpr size_t C3_OFF = C2_OFF + C_SZ;
static constexpr size_t KP_OFF = C3_OFF + C_SZ;            static constexpr size_t KP_SZ = (size_t)BATCH * KATT * 4;
static constexpr size_t BAR_OFF = KP_OFF + KP_SZ;          static constexpr size_t BAR_SZ = 256;
static constexpr size_t H3_OFF = BAR_OFF + BAR_SZ;         static constexpr size_t H3_SZ = (size_t)HROWS * KMDNPAD * 2;
static constexpr size_t ZERO_BYTES = H3_OFF + H3_SZ - ZERO_OFF;

// ---------------- helpers ----------------
__device__ __forceinline__ v16h load16(const _Float16* p) {
  h8 lo = *(const h8*)p;           // K +0..7
  h8 hi = *(const h8*)(p + 16);    // K +16..23
  return __builtin_shufflevector(lo, hi, 0,1,2,3,4,5,6,7,8,9,10,11,12,13,14,15);
}

__device__ __forceinline__ v8f wmma_f16(v16h a, v16h b, v8f c) {
  return __builtin_amdgcn_wmma_f32_16x16x32_f16(false, a, false, b, (short)0, c, false, false);
}

__device__ __forceinline__ float sigm(float x) { return 1.0f / (1.0f + __expf(-x)); }

__device__ __forceinline__ void gsync(unsigned int* cnt, unsigned int* gen) {
  __syncthreads();
  if (threadIdx.x == 0) {
    __hip_atomic_fetch_add(cnt, 1u, __ATOMIC_ACQ_REL, __HIP_MEMORY_SCOPE_AGENT);
    const unsigned int target = (*gen + 1u) * (unsigned int)NWG;
    while (__hip_atomic_load(cnt, __ATOMIC_ACQUIRE, __HIP_MEMORY_SCOPE_AGENT) < target)
      __builtin_amdgcn_s_sleep(1);
  }
  ++(*gen);
  __syncthreads();
}

// one 16-col strip (both M-tiles) of  Z[32,1600] = A[32,Kpad] @ Wpacked
__device__ __forceinline__ void gemm_phase(const _Float16* __restrict__ Wp,
                                           const _Float16* __restrict__ A,
                                           float* __restrict__ Z,
                                           int KT, int Kpad, int g, int lane) {
  if (g >= NT_G) return;
  const int nt  = g;
  const int row = lane & 15;
  const int ko  = (lane < 16) ? 0 : 8;   // A-frag K sub-offset per ISA layout
  v8f acc0 = {0.f,0.f,0.f,0.f,0.f,0.f,0.f,0.f};
  v8f acc1 = {0.f,0.f,0.f,0.f,0.f,0.f,0.f,0.f};
  for (int kt = 0; kt < KT; ++kt) {
    const _Float16* bp = Wp + ((size_t)(kt * NT_G + nt) * 32 + lane) * 16;
    v16h bf = *(const v16h*)bp;
    if (kt + 1 < KT)
      __builtin_prefetch((const void*)(Wp + ((size_t)((kt + 1) * NT_G + nt) * 32 + lane) * 16), 0, 0);
    const _Float16* ap = A + (size_t)row * Kpad + kt * 32 + ko;
    v16h a0 = load16(ap);
    v16h a1 = load16(ap + (size_t)16 * Kpad);
    acc0 = wmma_f16(a0, bf, acc0);
    acc1 = wmma_f16(a1, bf, acc1);
  }
  const int col   = nt * 16 + (lane & 15);
  const int rbase = (lane < 16) ? 0 : 8;
#pragma unroll
  for (int m = 0; m < 8; ++m) {
    Z[(size_t)(rbase + m) * GATES + col]      = acc0[m];
    Z[(size_t)(rbase + m + 16) * GATES + col] = acc1[m];
  }
}

// peephole LSTM gate nonlinearity; writes h (f16) to two destinations
__device__ __forceinline__ void ew_phase(const float* __restrict__ Z,
                                         const float* __restrict__ bias,
                                         const float* __restrict__ p,
                                         float* __restrict__ c,
                                         _Float16* __restrict__ h1d, int s1, int o1,
                                         _Float16* __restrict__ h2d, int s2, int o2,
                                         int gtid) {
  for (int j = gtid; j < BATCH * UNITS; j += TOTTHR) {
    const int b = j / UNITS, u = j - b * UNITS;
    const float* zr = Z + (size_t)b * GATES + u;
    float cc = c[j];
    float zi = zr[0]              + bias[u];
    float zf = zr[UNITS]          + bias[UNITS + u];
    float zg = zr[2 * UNITS]      + bias[2 * UNITS + u];
    float zo = zr[3 * UNITS]      + bias[3 * UNITS + u];
    float ig = sigm(zi + p[u] * cc);
    float fg = sigm(zf + p[UNITS + u] * cc);
    float cn = fg * cc + ig * tanhf(zg);
    float og = sigm(zo + p[2 * UNITS + u] * cn);
    float hh = og * tanhf(cn);
    c[j] = cn;
    _Float16 hf = (_Float16)hh;
    h1d[(size_t)b * s1 + o1 + u] = hf;
    h2d[(size_t)b * s2 + o2 + u] = hf;
  }
}

// ---------------- kernels ----------------
__global__ void zero_ws_kernel(char* ws) {
  uint4* p = (uint4*)(ws + ZERO_OFF);
  const size_t n = ZERO_BYTES / 16;
  for (size_t i = blockIdx.x * (size_t)blockDim.x + threadIdx.x; i < n;
       i += (size_t)gridDim.x * blockDim.x)
    p[i] = make_uint4(0u, 0u, 0u, 0u);
}

// pack f32 weights -> f16 WMMA B-fragment blobs ([kt][nt][lane][16 halves])
__global__ void pack_weights_kernel(char* ws,
    const float* __restrict__ Wx0, const float* __restrict__ Wh0,
    const float* __restrict__ Wx1, const float* __restrict__ Wh1,
    const float* __restrict__ Wx2, const float* __restrict__ Wh2,
    const float* __restrict__ Wmdn) {
  const long N1 = (long)KT1 * NT_G * 512;
  const long N2 = (long)KT23 * NT_G * 512;
  const long NM = (long)KTM * NTM * 512;
  const long total = N1 + 2 * N2 + NM;
  _Float16* out = (_Float16*)ws;     // all four regions are contiguous from offset 0
  const long stride = (long)gridDim.x * blockDim.x;
  for (long e = blockIdx.x * (long)blockDim.x + threadIdx.x; e < total; e += stride) {
    long le; int NT, region = 3, Kx = 0, Ktot = 0;
    const float *Wx = nullptr, *Wh = nullptr;
    if (e < N1)                { le = e;               NT = NT_G; region = 0; Wx = Wx0; Wh = Wh0; Kx = 76;  Ktot = 476; }
    else if (e < N1 + N2)      { le = e - N1;          NT = NT_G; region = 1; Wx = Wx1; Wh = Wh1; Kx = 476; Ktot = 876; }
    else if (e < N1 + 2 * N2)  { le = e - N1 - N2;     NT = NT_G; region = 2; Wx = Wx2; Wh = Wh2; Kx = 476; Ktot = 876; }
    else                       { le = e - N1 - 2 * N2; NT = NTM; }
    const long tile = le >> 9;
    const int kt = (int)(tile / NT), nt = (int)(tile % NT);
    const int lane = (int)((le >> 4) & 31), h = (int)(le & 15);
    const int krow = kt * 32 + ((lane & 16) ? 8 : 0) + (h & 7) + ((h & 8) ? 16 : 0);   // B layout: lanes 0-15 K+0..7,16..23; 16-31 K+8..15,24..31
    const int ncol = nt * 16 + (lane & 15);
    float v = 0.f;
    if (region == 3) {
      if (krow < UNITS && ncol < YCOLS) v = Wmdn[(size_t)krow * YCOLS + ncol];
    } else {
      if (krow < Kx)        v = Wx[(size_t)krow * GATES + ncol];
      else if (krow < Ktot) v = Wh[(size_t)(krow - Kx) * GATES + ncol];
    }
    out[e] = (_Float16)v;
  }
}

// persistent RNN scan: 13 WGs * 8 wave32, device-wide barriers between phases
__global__ void __launch_bounds__(NTHR)
rnn_scan_kernel(char* ws,
                const float* __restrict__ strokes, const int* __restrict__ chars,
                const int* __restrict__ lens,
                const float* __restrict__ b0, const float* __restrict__ p0,
                const float* __restrict__ b1, const float* __restrict__ p1,
                const float* __restrict__ b2, const float* __restrict__ p2,
                const float* __restrict__ W_att, const float* __restrict__ b_att) {
  _Float16* A1 = (_Float16*)(ws + A1_OFF);
  _Float16* A2 = (_Float16*)(ws + A2_OFF);
  _Float16* A3 = (_Float16*)(ws + A3_OFF);
  const _Float16* W1P = (const _Float16*)(ws + W1P_OFF);
  const _Float16* W2P = (const _Float16*)(ws + W2P_OFF);
  const _Float16* W3P = (const _Float16*)(ws + W3P_OFF);
  float* Z  = (float*)(ws + Z_OFF);
  float* C1 = (float*)(ws + C1_OFF);
  float* C2 = (float*)(ws + C2_OFF);
  float* C3 = (float*)(ws + C3_OFF);
  float* KAP = (float*)(ws + KP_OFF);
  _Float16* H3 = (_Float16*)(ws + H3_OFF);
  unsigned int* bar = (unsigned int*)(ws + BAR_OFF);

  const int tid  = threadIdx.x;
  const int lane = tid & 31;
  const int g    = blockIdx.x * (NTHR / 32) + (tid >> 5);
  const int gtid = blockIdx.x * NTHR + tid;
  unsigned int gen = 0;

  __shared__ float att_s[BATCH * 3 * KATT];   // [32][30]
  __shared__ float w_s[BATCH * NCHARS];       // [32][73]

  // prologue: x_0 into A1 (w/h regions already zeroed)
  if (blockIdx.x == 0) {
    for (int j = tid; j < BATCH * 3; j += NTHR) {
      int b = j / 3, d = j - b * 3;
      A1[(size_t)b * K1PAD + d] = (_Float16)strokes[((size_t)b * TSTEPS) * 3 + d];
    }
  }
  gsync(bar, &gen);

  for (int t = 0; t < TSTEPS; ++t) {
    // ----- layer 1 gates -----
    gemm_phase(W1P, A1, Z, KT1, K1PAD, g, lane);
    gsync(bar, &gen);
    ew_phase(Z, b0, p0, C1, A2, K23PAD, 76, A1, K1PAD, 76, gtid);  // h1 -> A2 (this step), A1 (next step)
    gsync(bar, &gen);

    // ----- attention + window + x staging (WG0) -----
    if (blockIdx.x == 0) {
      for (int j = tid; j < BATCH * 3 * KATT; j += NTHR) {      // att = h1 @ W_att + b_att
        int b = j / (3 * KATT), n = j - b * 3 * KATT;
        const _Float16* h1 = A2 + (size_t)b * K23PAD + 76;
        float s = b_att[n];
        for (int k = 0; k < UNITS; ++k) s += (float)h1[k] * W_att[(size_t)k * 3 * KATT + n];
        att_s[j] = s;
      }
      __syncthreads();
      for (int j = tid; j < BATCH * NCHARS; j += NTHR) w_s[j] = 0.f;
      __syncthreads();
      if (tid < BATCH) {
        const int b = tid;
        float al[KATT], be[KATT], kap[KATT];
#pragma unroll
        for (int k = 0; k < KATT; ++k) {
          al[k] = __expf(att_s[b * 3 * KATT + k]);
          be[k] = __expf(att_s[b * 3 * KATT + KATT + k]);
          float nk = KAP[b * KATT + k] + __expf(att_s[b * 3 * KATT + 2 * KATT + k]);
          kap[k] = nk; KAP[b * KATT + k] = nk;
        }
        int len = lens[b]; if (len > UC) len = UC;
        for (int u = 0; u < len; ++u) {
          float phi = 0.f;
#pragma unroll
          for (int k = 0; k < KATT; ++k) {
            float d = kap[k] - (float)u;
            phi += al[k] * __expf(-be[k] * d * d);
          }
          int cch = chars[b * UC + u];
          if (cch >= 0 && cch < NCHARS) w_s[b * NCHARS + cch] += phi;
        }
      }
      __syncthreads();
      for (int j = tid; j < BATCH * NCHARS; j += NTHR) {        // scatter window
        int b = j / NCHARS, cc = j - b * NCHARS;
        _Float16 v = (_Float16)w_s[j];
        A1[(size_t)b * K1PAD + 3 + cc] = v;     // for next step's layer1
        A2[(size_t)b * K23PAD + 3 + cc] = v;
        A3[(size_t)b * K23PAD + 3 + cc] = v;
      }
      for (int j = tid; j < BATCH * 3; j += NTHR) {             // x staging
        int b = j / 3, d = j - b * 3;
        float xt = strokes[((size_t)b * TSTEPS + t) * 3 + d];
        A2[(size_t)b * K23PAD + d] = (_Float16)xt;
        A3[(size_t)b * K23PAD + d] = (_Float16)xt;
        float xn = (t + 1 < TSTEPS) ? strokes[((size_t)b * TSTEPS + t + 1) * 3 + d] : 0.f;
        A1[(size_t)b * K1PAD + d] = (_Float16)xn;
      }
    }
    gsync(bar, &gen);

    // ----- layer 2 -----
    gemm_phase(W2P, A2, Z, KT23, K23PAD, g, lane);
    gsync(bar, &gen);
    ew_phase(Z, b1, p1, C2, A3, K23PAD, 76, A2, K23PAD, 476, gtid);  // h2 -> A3 (this step), A2 (next step)
    gsync(bar, &gen);

    // ----- layer 3 -----
    gemm_phase(W3P, A3, Z, KT23, K23PAD, g, lane);
    gsync(bar, &gen);
    ew_phase(Z, b2, p2, C3, A3, K23PAD, 476,
             H3 + (size_t)t * BATCH * KMDNPAD, KMDNPAD, 0, gtid);    // h3 -> A3 (next step), H3 buffer
    gsync(bar, &gen);
  }
}

// y[12800,121] = H3[12800,416] @ Wmdn_packed + b_mdn  (WMMA), reordered to [B,T,121]
__global__ void mdn_gemm_kernel(const char* ws, const float* __restrict__ bmdn,
                                float* __restrict__ out) {
  const _Float16* H3 = (const _Float16*)(ws + H3_OFF);
  const _Float16* Wp = (const _Float16*)(ws + WMP_OFF);
  const int lane = threadIdx.x & 31;
  const int g = (int)((blockIdx.x * blockDim.x + threadIdx.x) >> 5);
  const int nwaves = (int)((gridDim.x * blockDim.x) >> 5);
  const int ntasks = (HROWS / 16) * NTM;   // 800 * 8
  for (int task = g; task < ntasks; task += nwaves) {
    const int mt = task >> 3, nt = task & 7;
    v8f acc = {0.f,0.f,0.f,0.f,0.f,0.f,0.f,0.f};
    const int row = mt * 16 + (lane & 15);
    const int ko  = (lane < 16) ? 0 : 8;
    for (int kt = 0; kt < KTM; ++kt) {
      v16h a = load16(H3 + (size_t)row * KMDNPAD + kt * 32 + ko);
      v16h b = *(const v16h*)(Wp + ((size_t)(kt * NTM + nt) * 32 + lane) * 16);
      acc = wmma_f16(a, b, acc);
    }
    const int col = nt * 16 + (lane & 15);
    if (col < YCOLS) {
      const int rb = mt * 16 + ((lane < 16) ? 0 : 8);
      const float bb = bmdn[col];
#pragma unroll
      for (int m = 0; m < 8; ++m) {
        const int h3r = rb + m;              // = t*32 + b
        const int tt = h3r >> 5, bb2 = h3r & 31;
        out[((size_t)bb2 * TSTEPS + tt) * YCOLS + col] = acc[m] + bb;
      }
    }
  }
}

// in-place MDN output transforms (softmax / exp / tanh / sigmoid)
__global__ void mdn_xform_kernel(float* __restrict__ out) {
  const int r = blockIdx.x * blockDim.x + threadIdx.x;
  if (r >= HROWS) return;
  float* y = out + (size_t)r * YCOLS;
  float mx = y[0];
#pragma unroll
  for (int i = 1; i < NMIX; ++i) mx = fmaxf(mx, y[i]);
  float e[NMIX], s = 0.f;
#pragma unroll
  for (int i = 0; i < NMIX; ++i) { e[i] = __expf(y[i] - mx); s += e[i]; }
  const float inv = 1.f / s;
#pragma unroll
  for (int i = 0; i < NMIX; ++i) y[i] = e[i] * inv;
#pragma unroll
  for (int i = 3 * NMIX; i < 5 * NMIX; ++i) y[i] = __expf(y[i]);
#pragma unroll
  for (int i = 5 * NMIX; i < 6 * NMIX; ++i) y[i] = tanhf(y[i]);
  y[6 * NMIX] = 1.f / (1.f + __expf(-y[6 * NMIX]));
}

// ---------------- launcher ----------------
extern "C" void kernel_launch(void* const* d_in, const int* in_sizes, int n_in,
                              void* d_out, int out_size, void* d_ws, size_t ws_size,
                              hipStream_t stream) {
  (void)in_sizes; (void)n_in; (void)out_size; (void)ws_size;
  char* ws = (char*)d_ws;
  const float* strokes = (const float*)d_in[0];
  const int*   chars   = (const int*)d_in[1];
  const int*   lens    = (const int*)d_in[2];
  const float* Wx0 = (const float*)d_in[3];
  const float* Wh0 = (const float*)d_in[4];
  const float* b0  = (const float*)d_in[5];
  const float* p0  = (const float*)d_in[6];
  const float* Wx1 = (const float*)d_in[7];
  const float* Wh1 = (const float*)d_in[8];
  const float* b1  = (const float*)d_in[9];
  const float* p1  = (const float*)d_in[10];
  const float* Wx2 = (const float*)d_in[11];
  const float* Wh2 = (const float*)d_in[12];
  const float* b2  = (const float*)d_in[13];
  const float* p2  = (const float*)d_in[14];
  const float* W_att = (const float*)d_in[15];
  const float* b_att = (const float*)d_in[16];
  const float* Wmdn  = (const float*)d_in[17];
  const float* bmdn  = (const float*)d_in[18];
  float* out = (float*)d_out;

  zero_ws_kernel<<<512, 256, 0, stream>>>(ws);
  pack_weights_kernel<<<1024, 256, 0, stream>>>(ws, Wx0, Wh0, Wx1, Wh1, Wx2, Wh2, Wmdn);
  rnn_scan_kernel<<<NWG, NTHR, 0, stream>>>(ws, strokes, chars, lens,
                                            b0, p0, b1, p1, b2, p2, W_att, b_att);
  mdn_gemm_kernel<<<100, 256, 0, stream>>>(ws, bmdn, out);
  mdn_xform_kernel<<<(HROWS + 255) / 256, 256, 0, stream>>>(out);
}
